// E2EROIFusionHead_26259430048579
// MI455X (gfx1250) — compile-verified, matrix operand order checked by reading, not run on hardware
//
#include <hip/hip_runtime.h>
#include <math.h>

// ---------------- CDNA5 WMMA / TDM types ----------------
typedef __attribute__((ext_vector_type(16))) __bf16 v16bf;
typedef __attribute__((ext_vector_type(8)))  float  v8f;
typedef unsigned int v4u __attribute__((ext_vector_type(4)));
typedef int v4i __attribute__((ext_vector_type(4)));
typedef int v8i __attribute__((ext_vector_type(8)));

#define N_ROI   4096
#define N_PTS   64
#define NUM_REF 8

#define TDM_OK (__has_builtin(__builtin_amdgcn_tensor_load_to_lds) && \
                __has_builtin(__builtin_amdgcn_s_wait_tensorcnt))

__device__ __forceinline__ unsigned short f32_to_bf16_rne(float f) {
  unsigned int u = __float_as_uint(f);
  u += 0x7FFFu + ((u >> 16) & 1u);
  return (unsigned short)(u >> 16);
}
__device__ __forceinline__ __bf16 us_as_bf16(unsigned short u) {
  return __builtin_bit_cast(__bf16, u);
}
// 16-bit A/B fragment K-map (ISA 7.12.2): lanes 0-15 VGPR0..3 -> K0..7, VGPR4..7 -> K16..23;
// lanes 16-31 are +8 within each group.
__device__ __forceinline__ int wmma_kmap(int e, int half) {
  int vg = e >> 1, lo = e & 1;
  int k = (vg < 4) ? (vg * 2 + lo) : (16 + (vg - 4) * 2 + lo);
  return k + (half ? 8 : 0);
}

// ---------------- bf16 WMMA GEMM: C[m,n] = sum_k A[m,k]*W[n,k] (+bias)(+addend)(relu) ----------
// Block tile 128x32, 256 threads = 8 waves. Wave w owns M rows [w*16, w*16+16) and BOTH 16-col
// N tiles -> A fragment reused across 2 WMMAs per K step (2x matrix-op density vs 1 tile/wave).
__global__ __launch_bounds__(256) void gemm_bf16_wmma_kernel(
    const float* __restrict__ A, int lda, int acol,
    const float* __restrict__ W,          // N x K row-major
    const float* __restrict__ bias,
    const float* __restrict__ addend, int ldadd,
    float* __restrict__ C, int ldc,
    int M, int N, int K, int relu)
{
  __shared__ unsigned short sA[128][32];
  __shared__ unsigned short sB[32][32];
  int tid  = threadIdx.x;
  int wave = tid >> 5, lane = tid & 31;
  int half = lane >> 4, l16 = lane & 15;
  int bm = blockIdx.x * 128;
  int bn = blockIdx.y * 32;
  int waveM = wave * 16;
  v8f acc0 = {0.f, 0.f, 0.f, 0.f, 0.f, 0.f, 0.f, 0.f};
  v8f acc1 = {0.f, 0.f, 0.f, 0.f, 0.f, 0.f, 0.f, 0.f};
  for (int k0 = 0; k0 < K; k0 += 32) {
#pragma unroll
    for (int i = 0; i < 16; ++i) {              // 128*32 A elements
      int e = tid + i * 256;
      int r = e >> 5, c = e & 31;
      sA[r][c] = f32_to_bf16_rne(A[(size_t)(bm + r) * lda + acol + k0 + c]);
    }
#pragma unroll
    for (int i = 0; i < 4; ++i) {               // 32*32 W elements
      int e = tid + i * 256;
      int r = e >> 5, c = e & 31;
      sB[r][c] = f32_to_bf16_rne(W[(size_t)(bn + r) * K + k0 + c]);
    }
    if (k0 + 32 < K)                            // warm GL2/L0 for next K tile
      __builtin_prefetch(&A[(size_t)(bm + (tid & 127)) * lda + acol + k0 + 32], 0, 1);
    __syncthreads();
    v16bf av, bv0, bv1;
#pragma unroll
    for (int e = 0; e < 16; ++e) {
      int kk = wmma_kmap(e, half);
      av[e]  = us_as_bf16(sA[waveM + l16][kk]);
      bv0[e] = us_as_bf16(sB[l16][kk]);
      bv1[e] = us_as_bf16(sB[16 + l16][kk]);
    }
    acc0 = __builtin_amdgcn_wmma_f32_16x16x32_bf16(false, av, false, bv0,
                                                   (short)0, acc0, false, false);
    acc1 = __builtin_amdgcn_wmma_f32_16x16x32_bf16(false, av, false, bv1,
                                                   (short)0, acc1, false, false);
    __syncthreads();
  }
#pragma unroll
  for (int r = 0; r < 8; ++r) {                 // C layout: m = 8*half + r, n = lane%16
    int m  = bm + waveM + half * 8 + r;
    int n0 = bn + l16;
    int n1 = bn + 16 + l16;
    float v0 = acc0[r], v1 = acc1[r];
    if (bias)   { v0 += bias[n0]; v1 += bias[n1]; }
    if (addend) { v0 += addend[(size_t)m * ldadd + n0]; v1 += addend[(size_t)m * ldadd + n1]; }
    if (relu)   { v0 = fmaxf(v0, 0.f); v1 = fmaxf(v1, 0.f); }
    C[(size_t)m * ldc + n0] = v0;
    C[(size_t)m * ldc + n1] = v1;
  }
}

// ---------------- small-K projection: y[m,n] = sum_k (x[m,k]-sub[m%mod,k]) * W[n,k] + b[n] ------
__global__ void smallk_lin_kernel(const float* __restrict__ x, int K,
                                  const float* __restrict__ sub, int subMod,
                                  const float* __restrict__ W, const float* __restrict__ bias,
                                  float* __restrict__ y, int N, size_t MN)
{
  size_t idx = (size_t)blockIdx.x * 256 + threadIdx.x;
  if (idx >= MN) return;
  size_t m = idx / (size_t)N; int n = (int)(idx % (size_t)N);
  float acc = bias ? bias[n] : 0.f;
  for (int k = 0; k < K; ++k) {
    float xv = x[m * K + k];
    if (sub) xv -= sub[(m % subMod) * K + k];
    acc += xv * W[(size_t)n * K + k];
  }
  y[m * (size_t)N + n] = acc;
}

// ---------------- LayerNorm (+optional residual, +optional relu), strided/offset output --------
__global__ __launch_bounds__(128) void ln_kernel(
    const float* __restrict__ x, int ldx,
    const float* __restrict__ res, int ldres,
    const float* __restrict__ g, const float* __restrict__ b,
    float* __restrict__ y, int ldy, int ycol, int dim, int relu)
{
  __shared__ float buf[256];
  __shared__ float red[128];
  size_t m = blockIdx.x;
  int tid = threadIdx.x;
  float s = 0.f;
  for (int i = tid; i < dim; i += 128) {
    float v = x[m * (size_t)ldx + i];
    if (res) v += res[m * (size_t)ldres + i];
    buf[i] = v; s += v;
  }
  red[tid] = s; __syncthreads();
  for (int o = 64; o > 0; o >>= 1) { if (tid < o) red[tid] += red[tid + o]; __syncthreads(); }
  float mean = red[0] / dim; __syncthreads();
  float vs = 0.f;
  for (int i = tid; i < dim; i += 128) { float d0 = buf[i] - mean; vs += d0 * d0; }
  red[tid] = vs; __syncthreads();
  for (int o = 64; o > 0; o >>= 1) { if (tid < o) red[tid] += red[tid + o]; __syncthreads(); }
  float inv = rsqrtf(red[0] / dim + 1e-5f); __syncthreads();
  for (int i = tid; i < dim; i += 128) {
    float v = (buf[i] - mean) * inv * g[i] + b[i];
    if (relu) v = fmaxf(v, 0.f);
    y[m * (size_t)ldy + ycol + i] = v;
  }
}

// ---------------- elementwise helpers ----------------------------------------------------------
__global__ void add_inplace_kernel(float* __restrict__ a, const float* __restrict__ b, size_t n)
{ size_t i = (size_t)blockIdx.x * 256 + threadIdx.x; if (i < n) a[i] += b[i]; }

__global__ void all_empty_kernel(const unsigned char* __restrict__ mask, int* __restrict__ flags)
{
  int roi = blockIdx.x * 256 + threadIdx.x; if (roi >= N_ROI) return;
  int cnt = 0;
  for (int s = 0; s < N_PTS; ++s) cnt += (mask[(size_t)roi * N_PTS + s] != 0);
  flags[roi] = (cnt == N_PTS) ? 1 : 0;
}
__global__ void zero_rows_kernel(float* __restrict__ x, const int* __restrict__ flags,
                                 int E, size_t total)
{
  size_t idx = (size_t)blockIdx.x * 256 + threadIdx.x; if (idx >= total) return;
  size_t m = idx / (size_t)E; int roi = (int)(m % N_ROI);
  if (flags[roi]) x[idx] = 0.f;
}

// ---------------- bev -> fused[:,224:352] via Tensor Data Mover ---------------------------------
// One workgroup per 64-row tile: wave 0 issues TENSOR_LOAD_TO_LDS with a 2D D# (128 f32 per row,
// row stride 128, tile 128x64), waits on TENSORcnt, then the workgroup scatters the LDS tile into
// the 352-stride fused buffer. Plain-copy fallback keeps identical launch geometry.
__global__ __launch_bounds__(256) void bev_tdm_copy_kernel(const float* __restrict__ bev,
                                                           float* __restrict__ fused)
{
#if TDM_OK
  __shared__ float tile[64 * 128];     // only LDS object in this kernel -> lds_addr == 0
  int r0 = blockIdx.x * 64;
  if (threadIdx.x < 32) {              // single wave issues the DMA (EXEC ignored by tensor ops)
    unsigned long long ga = (unsigned long long)(const void*)(bev + (size_t)r0 * 128);
    v4u g0;
    g0[0] = 1u;                                            // count=1, user mode, no gather
    g0[1] = 0u;                                            // lds_addr
    g0[2] = (unsigned)(ga & 0xFFFFFFFFu);                  // global_addr[31:0]
    g0[3] = (unsigned)((ga >> 32) & 0x1FFFFFFu) | (2u << 30);  // global_addr[56:32] | type=2
    v8i g1;
    g1[0] = (int)(2u << 16);                               // data_size = 4B
    g1[1] = (int)(128u << 16);                             // tensor_dim0[15:0]=128
    g1[2] = (int)((32768u & 0xFFFFu) << 16);               // dim0 hi=0 | tensor_dim1 lo
    g1[3] = (int)((32768u >> 16) | (128u << 16));          // dim1 hi | tile_dim0=128
    g1[4] = 64;                                            // tile_dim1=64, tile_dim2=0
    g1[5] = 128;                                           // tensor_dim0_stride = 128
    g1[6] = 0; g1[7] = 0;
    v4i g2 = {0, 0, 0, 0};
    v4i g3 = {0, 0, 0, 0};
#if defined(__clang_major__) && (__clang_major__ >= 23)
    v8i g4 = {0, 0, 0, 0, 0, 0, 0, 0};
    __builtin_amdgcn_tensor_load_to_lds(g0, g1, g2, g3, g4, 0);
#else
    __builtin_amdgcn_tensor_load_to_lds(g0, g1, g2, g3, 0);
#endif
    __builtin_amdgcn_s_wait_tensorcnt(0);
  }
  __syncthreads();
  for (int i = threadIdx.x; i < 64 * 128; i += 256) {
    int r = i >> 7, ccol = i & 127;
    fused[(size_t)(r0 + r) * 352 + 224 + ccol] = tile[i];
  }
#else
  int r0 = blockIdx.x * 64;
  for (int i = threadIdx.x; i < 64 * 128; i += 256) {
    int r = i >> 7, ccol = i & 127;
    fused[(size_t)(r0 + r) * 352 + 224 + ccol] = bev[(size_t)(r0 + r) * 128 + ccol];
  }
#endif
}

// ---------------- pib masked attention: per (roi, head) -----------------------------------------
__global__ __launch_bounds__(64) void pib_attn_kernel(
    const float* __restrict__ qp, const float* __restrict__ kp, const float* __restrict__ vp,
    const unsigned char* __restrict__ mask, float* __restrict__ o, int E, int d)
{
  __shared__ float sk[N_PTS * 64];
  __shared__ float sv[N_PTS * 64];
  __shared__ float sq[NUM_REF * 64];
  __shared__ float sc[NUM_REF * N_PTS];
  int roi = blockIdx.x, h = blockIdx.y, s = threadIdx.x;
  const int col = h * d;
  for (int dd = 0; dd < d; ++dd) {
    sk[s * 64 + dd] = kp[((size_t)s * N_ROI + roi) * E + col + dd];
    sv[s * 64 + dd] = vp[((size_t)s * N_ROI + roi) * E + col + dd];
  }
  for (int i = s; i < NUM_REF * d; i += 64) {
    int l = i / d, dd = i % d;
    sq[l * 64 + dd] = qp[((size_t)l * N_ROI + roi) * E + col + dd];
  }
  __syncthreads();
  bool msk = (mask[(size_t)roi * N_PTS + s] != 0) && (s != 0);  // mask[:,0] forced False
  float scale = rsqrtf((float)d);
  for (int l = 0; l < NUM_REF; ++l) {
    float acc = 0.f;
    for (int dd = 0; dd < d; ++dd) acc += sq[l * 64 + dd] * sk[s * 64 + dd];
    sc[l * N_PTS + s] = msk ? -1e9f : acc * scale;
  }
  __syncthreads();
  if (s < NUM_REF) {
    int l = s;
    float mx = -1e30f;
    for (int j = 0; j < N_PTS; ++j) mx = fmaxf(mx, sc[l * N_PTS + j]);
    float den = 0.f;
    for (int j = 0; j < N_PTS; ++j) den += expf(sc[l * N_PTS + j] - mx);
    float inv = 1.f / den;
    for (int j = 0; j < N_PTS; ++j) sc[l * N_PTS + j] = expf(sc[l * N_PTS + j] - mx) * inv;
  }
  __syncthreads();
  for (int i = s; i < NUM_REF * d; i += 64) {
    int l = i / d, dd = i % d;
    float acc = 0.f;
    for (int j = 0; j < N_PTS; ++j) acc += sc[l * N_PTS + j] * sv[j * 64 + dd];
    o[((size_t)l * N_ROI + roi) * E + col + dd] = acc;
  }
}

// ---------------- sat attention: L=1, S=8, d=128, per (roi, head) -------------------------------
__global__ __launch_bounds__(128) void sat_attn_kernel(
    const float* __restrict__ qp, const float* __restrict__ kp, const float* __restrict__ vp,
    float* __restrict__ o)
{
  __shared__ float red[128];
  __shared__ float sc[8];
  int roi = blockIdx.x, h = blockIdx.y, tid = threadIdx.x;
  float q = qp[(size_t)roi * 256 + h * 128 + tid];
  for (int s = 0; s < 8; ++s) {
    float kv = kp[((size_t)s * N_ROI + roi) * 256 + h * 128 + tid];
    red[tid] = q * kv; __syncthreads();
    for (int w = 64; w > 0; w >>= 1) { if (tid < w) red[tid] += red[tid + w]; __syncthreads(); }
    if (tid == 0) sc[s] = red[0] * 0.088388347648f;   // 1/sqrt(128)
    __syncthreads();
  }
  float mx = -1e30f;
  for (int s = 0; s < 8; ++s) mx = fmaxf(mx, sc[s]);
  float p[8], den = 0.f;
  for (int s = 0; s < 8; ++s) { p[s] = expf(sc[s] - mx); den += p[s]; }
  float acc = 0.f;
  for (int s = 0; s < 8; ++s)
    acc += p[s] * vp[((size_t)s * N_ROI + roi) * 256 + h * 128 + tid];
  o[(size_t)roi * 256 + h * 128 + tid] = acc / den;
}

// ---------------- blt attention: L=S=512, n=8, d=128, per (l, nb, h) ----------------------------
__global__ __launch_bounds__(128) void blt_attn_kernel(
    const float* __restrict__ qp, const float* __restrict__ kp, const float* __restrict__ vp,
    float* __restrict__ o)
{
  __shared__ float sq[128];
  __shared__ float sc[512];
  __shared__ float red[128];
  int l = blockIdx.x, nb = blockIdx.y >> 1, h = blockIdx.y & 1, tid = threadIdx.x;
  sq[tid] = qp[((size_t)l * 8 + nb) * 256 + h * 128 + tid];
  __syncthreads();
  for (int s = tid; s < 512; s += 128) {
    const float* kr = kp + ((size_t)s * 8 + nb) * 256 + h * 128;
    float acc = 0.f;
    for (int d0 = 0; d0 < 128; ++d0) acc += sq[d0] * kr[d0];
    sc[s] = acc * 0.088388347648f;
  }
  __syncthreads();
  float mx = -1e30f;
  for (int s = tid; s < 512; s += 128) mx = fmaxf(mx, sc[s]);
  red[tid] = mx; __syncthreads();
  for (int w = 64; w > 0; w >>= 1) { if (tid < w) red[tid] = fmaxf(red[tid], red[tid + w]); __syncthreads(); }
  mx = red[0]; __syncthreads();
  float den = 0.f;
  for (int s = tid; s < 512; s += 128) den += expf(sc[s] - mx);
  red[tid] = den; __syncthreads();
  for (int w = 64; w > 0; w >>= 1) { if (tid < w) red[tid] += red[tid + w]; __syncthreads(); }
  den = red[0]; __syncthreads();
  for (int s = tid; s < 512; s += 128) sc[s] = expf(sc[s] - mx) / den;
  __syncthreads();
  float acc = 0.f;
  for (int s = 0; s < 512; ++s) acc += sc[s] * vp[((size_t)s * 8 + nb) * 256 + h * 128 + tid];
  o[((size_t)l * 8 + nb) * 256 + h * 128 + tid] = acc;
}

// ---------------- blt input: gather rib + box_params projection ---------------------------------
__global__ void blt_input_kernel(const float* __restrict__ rib, const float* __restrict__ bp,
                                 const float* __restrict__ bW, const float* __restrict__ bb,
                                 float* __restrict__ bf)
{
  int idx = blockIdx.x * 256 + threadIdx.x;     // 4096*256 total
  int t = idx >> 8, n = idx & 255;
  int l = t >> 3, nb = t & 7;
  float s = bb[n];
  for (int k = 0; k < 9; ++k) s += bp[(size_t)t * 9 + k] * bW[n * 9 + k];
  bf[(size_t)t * 256 + n] = rib[((size_t)nb * 512 + l) * 256 + n] + s;
}

// ---------------- cls/reg heads ----------------------------------------------------------------
__global__ __launch_bounds__(64) void head_kernel(
    const float* __restrict__ bf,
    const float* __restrict__ W1c, const float* __restrict__ gc, const float* __restrict__ bc,
    const float* __restrict__ W2c, const float* __restrict__ b2c,
    const float* __restrict__ W1r, const float* __restrict__ gr, const float* __restrict__ br,
    const float* __restrict__ W2r, const float* __restrict__ b2r,
    float* __restrict__ out)
{
  __shared__ float x[256];
  __shared__ float hc[64];
  __shared__ float hr[64];
  int t = blockIdx.x; int l = t >> 3; int b = t & 7; int o = threadIdx.x;
  for (int i = o; i < 256; i += 64) x[i] = bf[(size_t)t * 256 + i];
  __syncthreads();
  float ac = 0.f, ar = 0.f;
  for (int k = 0; k < 256; ++k) { float xv = x[k]; ac += W1c[o * 256 + k] * xv; ar += W1r[o * 256 + k] * xv; }
  hc[o] = fmaxf(ac * gc[o] + bc[o], 0.f);
  hr[o] = fmaxf(ar * gr[o] + br[o], 0.f);
  __syncthreads();
  if (o < 3) {
    float s = b2c[o];
    for (int k = 0; k < 64; ++k) s += W2c[o * 64 + k] * hc[k];
    out[(size_t)b * 1536 + (size_t)o * 512 + l] = s;                // cls (8,3,512)
  }
  if (o >= 32 && o < 39) {
    int oo = o - 32;
    float s = b2r[oo];
    for (int k = 0; k < 64; ++k) s += W2r[oo * 64 + k] * hr[k];
    out[12288 + (size_t)b * 3584 + (size_t)oo * 512 + l] = s;       // reg (8,7,512)
  }
}

// =================== host orchestration =========================================================
static inline unsigned gblk(size_t n) { return (unsigned)((n + 255) / 256); }

static inline void gemm(const float* A, int lda, int acol, const float* W,
                        const float* bias, const float* addend, int ldadd,
                        float* C, int ldc, int M, int N, int K, int relu, hipStream_t st)
{
  dim3 grid(M / 128, N / 32);
  gemm_bf16_wmma_kernel<<<grid, 256, 0, st>>>(A, lda, acol, W, bias, addend, ldadd,
                                              C, ldc, M, N, K, relu);
}
static inline void smallk(const float* x, int K, const float* sub, int subMod,
                          const float* W, const float* bias, float* y, int N, size_t MN,
                          hipStream_t st)
{
  smallk_lin_kernel<<<gblk(MN), 256, 0, st>>>(x, K, sub, subMod, W, bias, y, N, MN);
}
static inline void ln(const float* x, int ldx, const float* res, int ldres,
                      const float* g, const float* b, float* y, int ldy, int ycol,
                      int dim, int relu, size_t tokens, hipStream_t st)
{
  ln_kernel<<<(unsigned)tokens, 128, 0, st>>>(x, ldx, res, ldres, g, b, y, ldy, ycol, dim, relu);
}

// pib sub-dict leaves, sorted-key (JAX pytree) order:
// 0 aWi 1 aWo 2 abi 3 abo 4 f1W 5 f1b 6 f2W 7 f2b 8 n1b 9 n1g 10 n2b 11 n2g
// 12 oW 13 ob 14 obb 15 og 16 pcW 17 pcb 18 pfW 19 pfb 20 refW 21 refb
static void run_pib(const float* const* P, int base, int E, int C,
                    const float* ref, const float* c, const float* pc, const float* pf,
                    const unsigned char* mask, float* fused, int fcol,
                    float* Bkv, float* Bv, float* Bkp, float* const* S, int* flags,
                    hipStream_t st)
{
  const int Mq = NUM_REF * N_ROI;   // 32768
  const int Ms = N_PTS * N_ROI;     // 262144
  const float* aWi = P[base + 0];
  const float* abi = P[base + 2];
  // q = (ref - c) @ refW^T + refb
  smallk(ref, 3, c, N_ROI, P[base + 20], P[base + 21], S[0], E, (size_t)Mq * E, st);
  // kv = (pc - c) @ pcW^T + pcb
  smallk(pc, 3, c, N_ROI, P[base + 16], P[base + 17], Bkv, E, (size_t)Ms * E, st);
  // v = pf @ pfW^T + pfb
  if (C >= 32) gemm(pf, C, 0, P[base + 18], P[base + 19], nullptr, 0, Bv, E, Ms, E, C, 0, st);
  else         smallk(pf, C, nullptr, 1, P[base + 18], P[base + 19], Bv, E, (size_t)Ms * E, st);
  // value input = kv + v
  add_inplace_kernel<<<gblk((size_t)Ms * E), 256, 0, st>>>(Bv, Bkv, (size_t)Ms * E);
  // MHA in-projections
  gemm(S[0], E, 0, aWi,                       abi,         nullptr, 0, S[1], E, Mq, E, E, 0, st); // qp
  gemm(Bkv,  E, 0, aWi + (size_t)E * E,       abi + E,     nullptr, 0, Bkp,  E, Ms, E, E, 0, st); // kp
  gemm(Bv,   E, 0, aWi + 2 * (size_t)E * E,   abi + 2 * E, nullptr, 0, Bkv,  E, Ms, E, E, 0, st); // vp (reuse Bkv)
  pib_attn_kernel<<<dim3(N_ROI, 2), 64, 0, st>>>(S[1], Bkp, Bkv, mask, S[2], E, E / 2);
  // out projection + all_empty zeroing
  gemm(S[2], E, 0, P[base + 1], P[base + 3], nullptr, 0, S[3], E, Mq, E, E, 0, st);
  all_empty_kernel<<<gblk(N_ROI), 256, 0, st>>>(mask, flags);
  zero_rows_kernel<<<gblk((size_t)Mq * E), 256, 0, st>>>(S[3], flags, E, (size_t)Mq * E);
  // rf = LN(af + q); FFN; LN; out lin; LN+relu -> fused columns
  ln(S[3], E, S[0], E, P[base + 9], P[base + 8], S[4], E, 0, E, 0, (size_t)Mq, st);
  gemm(S[4], E, 0, P[base + 4], P[base + 5], nullptr, 0, S[5], E, Mq, E, E, 1, st);
  gemm(S[5], E, 0, P[base + 6], P[base + 7], nullptr, 0, S[6], E, Mq, E, E, 0, st);
  ln(S[6], E, S[4], E, P[base + 11], P[base + 10], S[7], E, 0, E, 0, (size_t)Mq, st);
  gemm(S[7], E, 0, P[base + 12], P[base + 13], nullptr, 0, S[5], E, Mq, E, E, 0, st);
  ln(S[5], E, nullptr, 0, P[base + 15], P[base + 14], fused, 352, fcol, E, 1, (size_t)Mq, st);
}

extern "C" void kernel_launch(void* const* d_in, const int* in_sizes, int n_in,
                              void* d_out, int out_size, void* d_ws, size_t ws_size,
                              hipStream_t stream)
{
  (void)in_sizes; (void)out_size; (void)ws_size;
  if (n_in < 128) return;
  // params leaves 0..115 (nested dicts flattened alphabetically, JAX pytree order):
  //   box 0..19 | clsW1 20 clsW2 21 cls_b 22 cls_g 23 clsb2 24 |
  //   pib_points 25.. pib_xc3 47.. pib_xc4 69.. |
  //   regW1 91 regW2 92 reg_b 93 reg_g 94 regb2 95 | rib 96..115
  const float* P[116];
  for (int i = 0; i < 116; ++i) P[i] = (const float*)d_in[i];
  const float* ref = (const float*)d_in[116];
  const float* pc_pts = (const float*)d_in[117];
  const float* pc_c3  = (const float*)d_in[118];
  const float* pc_c4  = (const float*)d_in[119];
  const float* pf_pts = (const float*)d_in[120];
  const float* pf_c3  = (const float*)d_in[121];
  const float* pf_c4  = (const float*)d_in[122];
  const float* bev    = (const float*)d_in[123];
  const float* boxp   = (const float*)d_in[124];
  const unsigned char* m_pts = (const unsigned char*)d_in[125];
  const unsigned char* m_c3  = (const unsigned char*)d_in[126];
  const unsigned char* m_c4  = (const unsigned char*)d_in[127];
  const float* c = ref + (size_t)7 * N_ROI * 3;   // ref[-1]

  // ---- workspace bump allocator ----
  float* ws = (float*)d_ws;
  size_t off = 0;
  auto alloc = [&](size_t n) { float* p = ws + off; off += n; return p; };
  float* Bfused = alloc((size_t)32768 * 352);
  float* Bkv = alloc((size_t)262144 * 128);
  float* Bv  = alloc((size_t)262144 * 128);
  float* Bkp = alloc((size_t)262144 * 128);
  float* S[8];
  for (int i = 0; i < 8; ++i) S[i] = alloc((size_t)32768 * 128);
  int* flags = (int*)alloc(4096);

  // ---- the three point-in-box blocks -> fused[:, 0:224] ----
  run_pib(P, 25, 32, 2,   ref, c, pc_pts, pf_pts, m_pts, Bfused, 0,  Bkv, Bv, Bkp, S, flags, stream);
  run_pib(P, 47, 64, 64,  ref, c, pc_c3,  pf_c3,  m_c3,  Bfused, 32, Bkv, Bv, Bkp, S, flags, stream);
  run_pib(P, 69, 128, 128,ref, c, pc_c4,  pf_c4,  m_c4,  Bfused, 96, Bkv, Bv, Bkp, S, flags, stream);
  // bev -> fused[:, 224:352] via TDM (tile 64 rows x 128 cols per workgroup)
  bev_tdm_copy_kernel<<<32768 / 64, 256, 0, stream>>>(bev, Bfused);

  // ---- sat (rib) ----
  const int RB = 96;
  smallk(ref, 3, c, N_ROI, P[RB + 18], P[RB + 19], Bkv, 768, (size_t)32768 * 768, stream);  // r
  gemm(Bfused, 352, 0, P[RB + 16], P[RB + 17], Bkv, 768, Bv, 768, 32768, 768, 352, 0, stream); // qkv = fused@qW + qb + r
  const float* sWi = P[RB + 0]; const float* sbi = P[RB + 2];
  float* kp_s = Bkp;
  float* vp_s = Bkp + (size_t)32768 * 256;
  gemm(Bv + (size_t)7 * N_ROI * 768, 768, 0,   sWi,                   sbi,       nullptr, 0, S[0], 256, 4096,  256, 256, 0, stream); // qp
  gemm(Bv, 768, 256, sWi + 256 * 256,          sbi + 256,             nullptr, 0, kp_s, 256, 32768, 256, 256, 0, stream);            // kp
  gemm(Bv, 768, 512, sWi + 2 * 256 * 256,      sbi + 512,             nullptr, 0, vp_s, 256, 32768, 256, 256, 0, stream);            // vp
  sat_attn_kernel<<<dim3(N_ROI, 2), 128, 0, stream>>>(S[0], kp_s, vp_s, S[1]);
  gemm(S[1], 256, 0, P[RB + 1], P[RB + 3], nullptr, 0, S[2], 256, 4096, 256, 256, 0, stream);
  ln(S[2], 256, Bv + (size_t)7 * N_ROI * 768, 768, P[RB + 9], P[RB + 8], S[3], 256, 0, 256, 0, 4096, stream);
  gemm(S[3], 256, 0, P[RB + 4], P[RB + 5], nullptr, 0, S[4], 256, 4096, 256, 256, 1, stream);
  gemm(S[4], 256, 0, P[RB + 6], P[RB + 7], nullptr, 0, S[5], 256, 4096, 256, 256, 0, stream);
  ln(S[5], 256, S[3], 256, P[RB + 11], P[RB + 10], S[2], 256, 0, 256, 0, 4096, stream);
  gemm(S[2], 256, 0, P[RB + 12], P[RB + 13], nullptr, 0, S[4], 256, 4096, 256, 256, 0, stream);
  ln(S[4], 256, nullptr, 0, P[RB + 15], P[RB + 14], S[6], 256, 0, 256, 1, 4096, stream);  // rib -> S[6]

  // ---- blt (box) ----
  // box leaves: 0 aWi 1 aWo 2 abi 3 abo 4 bW 5 bb 6 f1W 7 f1b 8 f2W 9 f2b
  //             10 n1b 11 n1g 12 n2b 13 n2g 14 oW 15 ob 16 obb 17 og 18 qW 19 qb
  blt_input_kernel<<<gblk((size_t)4096 * 256), 256, 0, stream>>>(S[6], boxp, P[4], P[5], S[7]);
  gemm(S[7], 256, 0, P[18], P[19], nullptr, 0, Bkp, 768, 4096, 768, 256, 0, stream);       // qkv2
  const float* bWi = P[0]; const float* bbi = P[2];
  gemm(Bkp, 768, 0,   bWi,                 bbi,       nullptr, 0, S[0], 256, 4096, 256, 256, 0, stream);
  gemm(Bkp, 768, 256, bWi + 256 * 256,     bbi + 256, nullptr, 0, S[1], 256, 4096, 256, 256, 0, stream);
  gemm(Bkp, 768, 512, bWi + 2 * 256 * 256, bbi + 512, nullptr, 0, S[2], 256, 4096, 256, 256, 0, stream);
  blt_attn_kernel<<<dim3(512, 16), 128, 0, stream>>>(S[0], S[1], S[2], S[3]);
  gemm(S[3], 256, 0, P[1], P[3], nullptr, 0, S[4], 256, 4096, 256, 256, 0, stream);
  ln(S[4], 256, S[7], 256, P[11], P[10], S[5], 256, 0, 256, 0, 4096, stream);
  gemm(S[5], 256, 0, P[6], P[7], nullptr, 0, S[0], 256, 4096, 256, 256, 1, stream);
  gemm(S[0], 256, 0, P[8], P[9], nullptr, 0, S[1], 256, 4096, 256, 256, 0, stream);
  ln(S[1], 256, S[5], 256, P[13], P[12], S[2], 256, 0, 256, 0, 4096, stream);
  gemm(S[2], 256, 0, P[14], P[15], nullptr, 0, S[3], 256, 4096, 256, 256, 0, stream);
  ln(S[3], 256, nullptr, 0, P[17], P[16], S[4], 256, 0, 256, 1, 4096, stream);

  // ---- heads ----
  head_kernel<<<4096, 64, 0, stream>>>(S[4],
      P[20], P[23], P[22], P[21], P[24],   // clsW1, cls_g, cls_b, clsW2, clsb2
      P[91], P[94], P[93], P[92], P[95],   // regW1, reg_g, reg_b, regW2, regb2
      (float*)d_out);
}
